// GRIDGRU_71064528879930
// MI455X (gfx1250) — matrix operand order repaired and checked by reading, not compiled, not run on hardware
//
#include <hip/hip_runtime.h>
#include <hip/hip_bf16.h>
#include <stdint.h>

// GridGRU on MI455X (gfx1250, wave32). bf16 WMMA 16x16x32 with f32 accum.
// Async global->LDS (ASYNCcnt) double-buffered pipeline; weight pre-transposed
// so both A and B tiles are contiguous async B128 copies (no LDS scatter).
// N=32, T=512, D=H=1024. Assumes ws_size >= ~400MB.

#define Nn 32
#define Tt 512
#define Dd 1024
#define Hh 1024
#define LDW 6144           // weight row stride (3*(D+H))
#define LDWT 2048          // transposed-weight row stride (D+H)
#define NB_SCAN 16

typedef __attribute__((ext_vector_type(16))) __bf16 v16bf;
typedef __attribute__((ext_vector_type(8)))  __bf16 v8bf;
typedef __attribute__((ext_vector_type(8)))  float  v8f;
typedef __attribute__((ext_vector_type(8)))  unsigned short u16x8;

__device__ __forceinline__ unsigned short f2bf(float f) {
    unsigned u = __float_as_uint(f);
    unsigned r = u + 0x7FFFu + ((u >> 16) & 1u);   // round-to-nearest-even
    return (unsigned short)(r >> 16);
}

__device__ __forceinline__ v16bf cat16(v8bf lo, v8bf hi) {
    return __builtin_shufflevector(lo, hi, 0,1,2,3,4,5,6,7,8,9,10,11,12,13,14,15);
}

// A fragment (16x32 bf16, MxK): lane<16: row=lane, K0..7 | K16..23;
// lane>=16: row=lane-16, K8..15 | K24..31.
__device__ __forceinline__ v16bf load_a_frag(const unsigned short* row_base, int lane) {
    int kb8 = (lane >> 4) << 3;                 // 0 or 8
    v8bf lo = *(const v8bf*)(row_base + kb8);
    v8bf hi = *(const v8bf*)(row_base + kb8 + 16);
    return cat16(lo, hi);
}

// B fragment (32x16 bf16, KxN) from K-transposed LDS tile Bt[n][k]:
// lane<16: col=lane, K0..7 | K16..23; lane>=16: col=lane-16, K8..15 | K24..31.
__device__ __forceinline__ v16bf load_b_frag(const unsigned short* col_base, int lane) {
    int kb8 = (lane >> 4) << 3;                 // 0 or 8
    v8bf lo = *(const v8bf*)(col_base + kb8);
    v8bf hi = *(const v8bf*)(col_base + kb8 + 16);
    return cat16(lo, hi);
}

// CDNA5 async global->LDS copy (16B per lane), tracked by ASYNCcnt.
__device__ __forceinline__ void async_b128(unsigned lds_off, const void* gptr) {
    asm volatile("global_load_async_to_lds_b128 %0, %1, off"
                 :: "v"(lds_off), "v"((unsigned long long)(uintptr_t)gptr)
                 : "memory");
}
__device__ __forceinline__ void wait_async4() {
    asm volatile("s_wait_asynccnt 4" ::: "memory");
}
__device__ __forceinline__ void wait_async5() {
    asm volatile("s_wait_asynccnt 5" ::: "memory");
}
__device__ __forceinline__ void wait_async0() {
    asm volatile("s_wait_asynccnt 0" ::: "memory");
}

// Fast activations: v_exp_f32 + v_rcp_f32 (avoid IEEE div chain / libm tanh).
__device__ __forceinline__ float fast_sigmoid(float x) {
    return __builtin_amdgcn_rcpf(1.0f + __expf(-x));
}
__device__ __forceinline__ float fast_tanh(float x) {
    float e = __expf(2.0f * x);                 // +inf for large x -> rcp=0 -> 1
    return 1.0f - 2.0f * __builtin_amdgcn_rcpf(e + 1.0f);
}

// ---------------------------------------------------------------------------
// Elementwise conversion kernels
// ---------------------------------------------------------------------------
__global__ void k_f32_to_bf16(const float* __restrict__ src,
                              unsigned short* __restrict__ dst, int n) {
    for (int i = blockIdx.x * blockDim.x + threadIdx.x; i < n; i += gridDim.x * blockDim.x)
        dst[i] = f2bf(src[i]);
}

// weight (2048 x 6144 f32, row-major) -> Wt (6144 x 2048 bf16, row-major)
__global__ void k_w_to_bf16t(const float* __restrict__ w,
                             unsigned short* __restrict__ wt, int n) {
    for (int i = blockIdx.x * blockDim.x + threadIdx.x; i < n; i += gridDim.x * blockDim.x) {
        int r = i / LDW, c = i - r * LDW;
        wt[(size_t)c * LDWT + r] = f2bf(w[i]);
    }
}

__global__ void k_init_h(const float* __restrict__ prev,
                         float* __restrict__ hws, unsigned short* __restrict__ hbf, int n) {
    for (int i = blockIdx.x * blockDim.x + threadIdx.x; i < n; i += gridDim.x * blockDim.x) {
        float v = prev[i];
        hws[i] = v;
        hbf[i] = f2bf(v);
    }
}

__global__ void k_copy_lasth(const float* __restrict__ hws, float* __restrict__ out, int n) {
    for (int i = blockIdx.x * blockDim.x + threadIdx.x; i < n; i += gridDim.x * blockDim.x)
        out[i] = hws[i];
}

// ---------------------------------------------------------------------------
// Big tiled GEMM: C[M x Ncols] = A[M x K] * Wt^T(b_row_off.., b_col_off..)
// Block = 256 threads = 8 waves (2 M-waves x 4 N-waves); tile 128x128, Ktile=32.
// Double-buffered async global->LDS pipeline (4 async B128 per thread / tile).
// A is a K-concat of A0 (k<Ksplit) and A1 (k>=Ksplit), both row stride 1024.
// MODE 0: gates = acc + bias  -> out0 (ldc 3072)
// MODE 1: s = sigmoid(acc + bias); col<1024 -> uds; else rx_bf = bf16(s * x)
// MODE 2: hc = tanh(acc + bias); out = x + u*(hc - x)
// ---------------------------------------------------------------------------
template<int MODE>
__global__ __launch_bounds__(256) void gemm128(
    const unsigned short* __restrict__ A0,
    const unsigned short* __restrict__ A1,
    int Ksplit,
    const unsigned short* __restrict__ Wt,
    int b_row_off, int b_col_off, int K,
    const float* __restrict__ bias, int bias_off,
    float* __restrict__ out0,
    const float* __restrict__ xf,
    unsigned short* __restrict__ rx_bf,
    const float* __restrict__ uds_in)
{
    __shared__ alignas(16) unsigned short As[2][128][40];
    __shared__ alignas(16) unsigned short Bt[2][128][40];

    const int tid  = threadIdx.x;
    const int lane = tid & 31;
    const int wv   = tid >> 5;      // 0..7
    const int wm   = wv >> 2;       // 0..1  -> M offset wm*64
    const int wn   = wv & 3;        // 0..3  -> N offset wn*32
    const int rowbase = blockIdx.y * 128;
    const int colbase = blockIdx.x * 128;

    v8f acc[4][2] = {};

    auto issue_tile = [&](int kb, int buf) {
        // A tile 128x32: 2 async B128 per thread
        const int ar = tid >> 1;
        #pragma unroll
        for (int p = 0; p < 2; ++p) {
            const int ac = ((tid & 1) << 3) + (p << 4);
            const int gk = kb + ac;
            const unsigned short* src = (gk < Ksplit)
                ? (A0 + (size_t)(rowbase + ar) * 1024 + gk)
                : (A1 + (size_t)(rowbase + ar) * 1024 + (gk - Ksplit));
            async_b128((unsigned)(uintptr_t)&As[buf][ar][ac], src);
        }
        // B tile 32x128 straight from transposed weight (contiguous K runs)
        #pragma unroll
        for (int p = 0; p < 2; ++p) {
            const int n  = (tid >> 2) + (p << 6);
            const int k8 = (tid & 3) << 3;
            const unsigned short* src =
                Wt + (size_t)(b_col_off + colbase + n) * LDWT + b_row_off + kb + k8;
            async_b128((unsigned)(uintptr_t)&Bt[buf][n][k8], src);
        }
    };

    const int NT = K / 32;
    issue_tile(0, 0);
    for (int i = 0; i < NT; ++i) {
        const int cur = i & 1;
        if (i + 1 < NT) { issue_tile((i + 1) * 32, cur ^ 1); wait_async4(); }
        else            { wait_async0(); }
        __syncthreads();    // tile `cur` visible to all waves

        v16bf bfrag[2];
        #pragma unroll
        for (int nt = 0; nt < 2; ++nt)
            bfrag[nt] = load_b_frag(&Bt[cur][wn * 32 + nt * 16 + (lane & 15)][0], lane);
        #pragma unroll
        for (int mt = 0; mt < 4; ++mt) {
            v16bf afrag = load_a_frag(&As[cur][wm * 64 + mt * 16 + (lane & 15)][0], lane);
            #pragma unroll
            for (int nt = 0; nt < 2; ++nt)
                acc[mt][nt] = __builtin_amdgcn_wmma_f32_16x16x32_bf16(
                    false, afrag, false, bfrag[nt], (short)0, acc[mt][nt], false, false);
        }
        __syncthreads();    // all reads of `cur` done before it is re-filled
    }

    // ---- epilogue ----
    const int mlocal = (lane >> 4) << 3;    // 0 or 8
    #pragma unroll
    for (int mt = 0; mt < 4; ++mt) {
        #pragma unroll
        for (int nt = 0; nt < 2; ++nt) {
            const int gn  = colbase + wn * 32 + nt * 16 + (lane & 15);
            const int gm0 = rowbase + wm * 64 + mt * 16 + mlocal;
            const float b = bias[bias_off + gn];
            #pragma unroll
            for (int r = 0; r < 8; ++r) {
                const int gm = gm0 + r;
                float v = acc[mt][nt][r] + b;
                if (MODE == 0) {
                    out0[(size_t)gm * 3072 + gn] = v;
                } else if (MODE == 1) {
                    float s = fast_sigmoid(v);
                    if (gn < 1024) {
                        out0[(size_t)gm * 1024 + gn] = s;          // u_d
                    } else {
                        const int rc = gn - 1024;
                        float xv = xf[(size_t)gm * 1024 + rc];
                        rx_bf[(size_t)gm * 1024 + rc] = f2bf(s * xv);
                    }
                } else { // MODE == 2
                    float hc = fast_tanh(v);
                    float xv = xf[(size_t)gm * 1024 + gn];
                    float u  = uds_in[(size_t)gm * 1024 + gn];
                    out0[(size_t)gm * 1024 + gn] = xv + u * (hc - xv);
                }
            }
        }
    }
}

// ---------------------------------------------------------------------------
// Global (grid-wide) sense barrier for the persistent scan kernel
// ---------------------------------------------------------------------------
__device__ __forceinline__ void gbar(unsigned* cnt, volatile unsigned* gen, unsigned nb) {
    __syncthreads();
    if (threadIdx.x == 0) {
        __threadfence();
        unsigned g = *gen;
        unsigned old = atomicAdd(cnt, 1u);
        if (old == nb - 1u) {
            *cnt = 0u;
            __threadfence();
            atomicAdd((unsigned*)gen, 1u);
        } else {
            while (*gen == g) { __builtin_amdgcn_s_sleep(1); }
        }
        __threadfence();
    }
    __syncthreads();
}

// ---------------------------------------------------------------------------
// Persistent GRU scan. 16 blocks x 128 threads (4 waves). Per step:
//   phase1 (16 blocks): ur = sigmoid(g[:, :2048] + h @ W[1024:, 0:2048])
//                       u -> uws (f32); r -> rh_bf = bf16(r * h)
//   phase2 (8 blocks):  hc = tanh(g[:, 2048:] + rh @ W[1024:, 2048:3072])
//                       h' = h + u*(hc - h); update hws/hbf, stream ht_bf
// Async double-buffered tiles (5 async B128 per thread / tile).
// ---------------------------------------------------------------------------
__global__ __launch_bounds__(128) void scan_kernel(
    const unsigned short* __restrict__ Wt,
    const float* __restrict__ gates,
    unsigned short* __restrict__ hbf,
    float* __restrict__ hws,
    unsigned short* __restrict__ rhbf,
    float* __restrict__ uws,
    unsigned short* __restrict__ htbf,
    unsigned* bar_cnt, unsigned* bar_gen)
{
    __shared__ alignas(16) unsigned short As[2][32][40];
    __shared__ alignas(16) unsigned short Bt[2][128][40];

    const int tid  = threadIdx.x;
    const int lane = tid & 31;
    const int wv   = tid >> 5;      // 0..3 -> N offset wv*32
    const int blk  = blockIdx.x;    // 0..15
    const int mlocal = (lane >> 4) << 3;

    auto issue_tile = [&](const unsigned short* Abase, int wcol, int kb, int buf) {
        // A tile 32x32: 1 async B128 per thread
        async_b128((unsigned)(uintptr_t)&As[buf][tid >> 2][(tid & 3) << 3],
                   Abase + (size_t)(tid >> 2) * 1024 + kb + ((tid & 3) << 3));
        // B tile 32x128 from transposed weight, rows 1024+k
        #pragma unroll
        for (int p = 0; p < 4; ++p) {
            const int n  = (tid >> 2) + (p << 5);
            const int k8 = (tid & 3) << 3;
            async_b128((unsigned)(uintptr_t)&Bt[buf][n][k8],
                       Wt + (size_t)(wcol + n) * LDWT + 1024 + kb + k8);
        }
    };

    auto do_gemm = [&](const unsigned short* Abase, int wcol, v8f (&acc)[2][2]) {
        issue_tile(Abase, wcol, 0, 0);
        for (int i = 0; i < 32; ++i) {
            const int cur = i & 1;
            if (i + 1 < 32) { issue_tile(Abase, wcol, (i + 1) * 32, cur ^ 1); wait_async5(); }
            else            { wait_async0(); }
            __syncthreads();
            v16bf bfrag[2];
            #pragma unroll
            for (int nt = 0; nt < 2; ++nt)
                bfrag[nt] = load_b_frag(&Bt[cur][wv * 32 + nt * 16 + (lane & 15)][0], lane);
            #pragma unroll
            for (int mt = 0; mt < 2; ++mt) {
                v16bf afrag = load_a_frag(&As[cur][mt * 16 + (lane & 15)][0], lane);
                #pragma unroll
                for (int nt = 0; nt < 2; ++nt)
                    acc[mt][nt] = __builtin_amdgcn_wmma_f32_16x16x32_bf16(
                        false, afrag, false, bfrag[nt], (short)0, acc[mt][nt], false, false);
            }
            __syncthreads();
        }
    };

    for (int t = 0; t < Tt; ++t) {
        // ---------------- phase 1 ----------------
        {
            v8f acc[2][2] = {};
            do_gemm(hbf, blk * 128, acc);
            #pragma unroll
            for (int mt = 0; mt < 2; ++mt) {
                #pragma unroll
                for (int nt = 0; nt < 2; ++nt) {
                    const int gcol = blk * 128 + wv * 32 + nt * 16 + (lane & 15);
                    #pragma unroll
                    for (int r = 0; r < 8; ++r) {
                        const int m = mt * 16 + mlocal + r;
                        float g = gates[((size_t)m * Tt + t) * 3072 + gcol];
                        float s = fast_sigmoid(acc[mt][nt][r] + g);
                        if (gcol < 1024) {
                            uws[m * 1024 + gcol] = s;
                        } else {
                            const int rc = gcol - 1024;
                            rhbf[m * 1024 + rc] = f2bf(s * hws[m * 1024 + rc]);
                        }
                    }
                }
            }
        }
        gbar(bar_cnt, bar_gen, NB_SCAN);

        // ---------------- phase 2 ----------------
        if (blk < 8) {
            v8f acc[2][2] = {};
            do_gemm(rhbf, 2048 + blk * 128, acc);
            #pragma unroll
            for (int mt = 0; mt < 2; ++mt) {
                #pragma unroll
                for (int nt = 0; nt < 2; ++nt) {
                    const int c = blk * 128 + wv * 32 + nt * 16 + (lane & 15);
                    #pragma unroll
                    for (int r = 0; r < 8; ++r) {
                        const int m = mt * 16 + mlocal + r;
                        float pre = acc[mt][nt][r]
                                  + gates[((size_t)m * Tt + t) * 3072 + 2048 + c];
                        float hc   = fast_tanh(pre);
                        float hold = hws[m * 1024 + c];
                        float u    = uws[m * 1024 + c];
                        float hn   = hold + u * (hc - hold);
                        hws[m * 1024 + c] = hn;
                        unsigned short hb = f2bf(hn);
                        hbf[m * 1024 + c] = hb;
                        htbf[((size_t)m * Tt + t) * 1024 + c] = hb;
                    }
                }
            }
        }
        gbar(bar_cnt, bar_gen, NB_SCAN);
    }
}

// ---------------------------------------------------------------------------
// Host launcher
// ---------------------------------------------------------------------------
extern "C" void kernel_launch(void* const* d_in, const int* in_sizes, int n_in,
                              void* d_out, int out_size, void* d_ws, size_t ws_size,
                              hipStream_t stream) {
    (void)in_sizes; (void)n_in; (void)out_size; (void)ws_size;
    const float* x    = (const float*)d_in[0];   // (N,T,D)
    const float* prev = (const float*)d_in[1];   // (N,H)
    const float* w    = (const float*)d_in[2];   // (2048, 6144)
    const float* bias = (const float*)d_in[3];   // (6144,)
    float* out = (float*)d_out;

    char* ws = (char*)d_ws;
    size_t off = 0;
    auto take = [&](size_t bytes) -> char* {
        char* p = ws + off;
        off = (off + bytes + 255) & ~(size_t)255;
        return p;
    };
    unsigned short* x_bf  = (unsigned short*)take((size_t)Nn * Tt * Dd * 2);  // 33.5MB
    unsigned short* wt_bf = (unsigned short*)take((size_t)LDW * LDWT * 2);    // 25.2MB
    unsigned short* ht_bf = (unsigned short*)take((size_t)Nn * Tt * Hh * 2);  // 33.5MB
    unsigned short* rx_bf = (unsigned short*)take((size_t)Nn * Tt * Dd * 2);  // 33.5MB
    float*          gates = (float*)take((size_t)Nn * Tt * 3072 * 4);         // 201MB
    float*          uds   = (float*)take((size_t)Nn * Tt * Dd * 4);           // 67MB
    float*          hws   = (float*)take((size_t)Nn * Hh * 4);
    unsigned short* hbf   = (unsigned short*)take((size_t)Nn * Hh * 2);
    unsigned short* rhbf  = (unsigned short*)take((size_t)Nn * Hh * 2);
    float*          uws   = (float*)take((size_t)Nn * Hh * 4);
    unsigned*       bar   = (unsigned*)take(256);

    hipMemsetAsync(bar, 0, 256, stream);

    // bf16 conversions (weight transposed for contiguous-K async B panels)
    k_f32_to_bf16<<<2048, 256, 0, stream>>>(x, x_bf, Nn * Tt * Dd);
    k_w_to_bf16t<<<2048, 256, 0, stream>>>(w, wt_bf, 2048 * LDW);
    k_init_h<<<128, 256, 0, stream>>>(prev, hws, hbf, Nn * Hh);

    // gates = x @ W[0:1024, 0:3072] + bias[0:3072]
    gemm128<0><<<dim3(24, 128), 256, 0, stream>>>(
        x_bf, x_bf, 1024, wt_bf, /*row*/0, /*col*/0, /*K*/1024,
        bias, /*bias_off*/0, gates, nullptr, nullptr, nullptr);

    // sequential GRU scan over T
    scan_kernel<<<NB_SCAN, 128, 0, stream>>>(
        wt_bf, gates, hbf, hws, rhbf, uws, ht_bf, bar, bar + 1);

    // urd = sigmoid(bias[3072:5120] + [x,ht] @ W[:, 3072:5120]) -> uds, rx_bf
    gemm128<1><<<dim3(16, 128), 256, 0, stream>>>(
        x_bf, ht_bf, 1024, wt_bf, 0, 3072, 2048,
        bias, 3072, uds, x, rx_bf, nullptr);

    // h = x + ud*(tanh(bias[5120:] + [rd*x,ht] @ W[:, 5120:6144]) - x)
    gemm128<2><<<dim3(8, 128), 256, 0, stream>>>(
        rx_bf, ht_bf, 1024, wt_bf, 0, 5120, 2048,
        bias, 5120, out, x, nullptr, uds);

    // last_h
    k_copy_lasth<<<128, 256, 0, stream>>>(hws, out + (size_t)Nn * Tt * Dd, Nn * Hh);
}
